// Convolutional_61186104098954
// MI455X (gfx1250) — matrix operand order repaired and checked by reading, not compiled, not run on hardware
//
#include <hip/hip_runtime.h>
#include <stdint.h>

// Problem constants (match reference)
#define B_  16
#define S_  256
#define T_  128
#define D_  512
#define K_  3
#define L_  3
#define V_  32000
#define C2_ (2*D_)
#define CK_ (D_*K_)
#define R2_ 0.70710678118654752440f

typedef __attribute__((ext_vector_type(16))) _Float16 v16h;
typedef __attribute__((ext_vector_type(2)))  __fp16   fp16x2;
typedef __attribute__((ext_vector_type(8)))  float    v8f;

// Single-instruction f32x2 -> packed f16 (v_cvt_pk_rz_f16_f32).
__device__ __forceinline__ uint32_t pack2h(float a, float b) {
  union { fp16x2 v; uint32_t u; } r;
  r.v = __builtin_amdgcn_cvt_pkrtz(a, b);
  return r.u;
}

// ---------------------------------------------------------------------------
// WMMA GEMM: C[M,N] = A[M,K] * B + bias
//   A row-major (lda).  BTRANS==0 -> B row-major [K,N] (ldb)
//                       BTRANS==1 -> B^T supplied, row-major [N,K] (ldb)
//   Batched via gridDim.z with element strides sA/sB/sC.
//
// INVARIANTS (all call sites satisfy them): M % 128 == 0, N % 128 == 0,
// K % 32 == 0, all row pointers 16-byte aligned. No bounds checks.
//
// 256 threads = 8 waves; block computes a 128x128 C tile; K stepped by 32.
// Each wave owns a 64x32 sub-tile = 4x2 accumulators of 16x16 f32
// -> 8 v_wmma_f32_16x16x32_f16 per K-step.
// Staging: global float4 (b128) loads -> registers -> packed f16 in LDS;
// next tile's global loads are issued before the current tile's WMMAs so
// HBM latency hides under ds_load + wmma work.
// ---------------------------------------------------------------------------
template <int BTRANS>
__launch_bounds__(256, 2)
__global__ void wmma_gemm_kernel(const float* __restrict__ A, int lda, long long sA,
                                 const float* __restrict__ B, int ldb, long long sB,
                                 float* __restrict__ C, int ldc, long long sC,
                                 const float* __restrict__ bias,
                                 int M, int N, int K)
{
  (void)M; (void)N;
  __shared__ uint32_t lA[128][17];   // 128 rows x 32 f16 (16 packed uints), +1 pad
  __shared__ uint32_t lB[128][17];   // 128 cols x 32 f16 (K-major per column)

  const float* Ab = A + (long long)blockIdx.z * sA;
  const float* Bb = B + (long long)blockIdx.z * sB;
  float*       Cb = C + (long long)blockIdx.z * sC;

  const int tid  = threadIdx.x;
  const int lane = tid & 31;
  const int wave = tid >> 5;
  const int m0 = blockIdx.y * 128;
  const int n0 = blockIdx.x * 128;
  const int wm = (wave >> 2) * 64;   // wave row offset: 0 or 64
  const int wn = (wave & 3) * 32;    // wave col offset: 0,32,64,96
  const int l16  = lane & 15;
  const int half = lane >> 4;        // 0: lanes 0-15, 1: lanes 16-31

  v8f acc[4][2];
#pragma unroll
  for (int ti = 0; ti < 4; ++ti)
#pragma unroll
    for (int tj = 0; tj < 2; ++tj)
      acc[ti][tj] = (v8f){0.f,0.f,0.f,0.f,0.f,0.f,0.f,0.f};

  float4 aR[4], bR[4];

  // ---- staged-load helpers (registers <-> global / LDS) ----
  auto loadA = [&](int kk) {
#pragma unroll
    for (int it = 0; it < 4; ++it) {
      int i = tid + it * 256;          // 0..1023
      int r = i >> 3, cq = i & 7;      // row 0..127, float4 quad 0..7
      aR[it] = *reinterpret_cast<const float4*>(
          Ab + (long long)(m0 + r) * lda + kk + 4 * cq);
    }
  };
  auto storeA = [&]() {
#pragma unroll
    for (int it = 0; it < 4; ++it) {
      int i = tid + it * 256;
      int r = i >> 3, cq = i & 7;
      lA[r][2 * cq]     = pack2h(aR[it].x, aR[it].y);
      lA[r][2 * cq + 1] = pack2h(aR[it].z, aR[it].w);
    }
  };
  auto loadB = [&](int kk) {
    if (BTRANS) {
#pragma unroll
      for (int it = 0; it < 4; ++it) {
        int i = tid + it * 256;
        int nn = i >> 3, cq = i & 7;
        bR[it] = *reinterpret_cast<const float4*>(
            Bb + (long long)(n0 + nn) * ldb + kk + 4 * cq);
      }
    } else {
      // n-inner mapping: consecutive threads read consecutive n (coalesced rows)
#pragma unroll
      for (int it = 0; it < 2; ++it) {
        int i = tid + it * 256;        // 0..511
        int cu = i >> 5, nq = i & 31;  // k-pair 0..15, n quad 0..31
        int gk = kk + 2 * cu;
        bR[2 * it]     = *reinterpret_cast<const float4*>(
            Bb + (long long)gk * ldb + n0 + 4 * nq);
        bR[2 * it + 1] = *reinterpret_cast<const float4*>(
            Bb + (long long)(gk + 1) * ldb + n0 + 4 * nq);
      }
    }
  };
  auto storeB = [&]() {
    if (BTRANS) {
#pragma unroll
      for (int it = 0; it < 4; ++it) {
        int i = tid + it * 256;
        int nn = i >> 3, cq = i & 7;
        lB[nn][2 * cq]     = pack2h(bR[it].x, bR[it].y);
        lB[nn][2 * cq + 1] = pack2h(bR[it].z, bR[it].w);
      }
    } else {
#pragma unroll
      for (int it = 0; it < 2; ++it) {
        int i = tid + it * 256;
        int cu = i >> 5, nq = i & 31;
        int nn0 = 4 * nq;
        lB[nn0 + 0][cu] = pack2h(bR[2 * it].x, bR[2 * it + 1].x);
        lB[nn0 + 1][cu] = pack2h(bR[2 * it].y, bR[2 * it + 1].y);
        lB[nn0 + 2][cu] = pack2h(bR[2 * it].z, bR[2 * it + 1].z);
        lB[nn0 + 3][cu] = pack2h(bR[2 * it].w, bR[2 * it + 1].w);
      }
    }
  };

  loadA(0);
  loadB(0);

  for (int k0 = 0; k0 < K; k0 += 32) {
    storeA();
    storeB();
    __syncthreads();

    // Issue next tile's HBM loads before consuming this tile.
    if (k0 + 32 < K) { loadA(k0 + 32); loadB(k0 + 32); }
    // Prefetch one further tile of A (emits global_prefetch_b8).
    if (k0 + 64 < K) {
      __builtin_prefetch(Ab + (long long)(m0 + (tid >> 3)) * lda + k0 + 64 + 4 * (tid & 7), 0, 1);
    }

    // Build fragments per CDNA5 ISA 7.12.2 layouts (wave32).
    // A 16x32 f16: lane half selects kbase (0/8); vgpr0-3 = K=kbase..kbase+7,
    //              vgpr4-7 = K=16+kbase..16+kbase+7.
    v16h afr[4], bfr[2];
#pragma unroll
    for (int ti = 0; ti < 4; ++ti) {
      int r   = wm + ti * 16 + l16;
      int kbu = half * 4;
      union { v16h v; uint32_t u[8]; } fa;
#pragma unroll
      for (int j = 0; j < 4; ++j) fa.u[j]     = lA[r][kbu + j];
#pragma unroll
      for (int j = 0; j < 4; ++j) fa.u[4 + j] = lA[r][8 + kbu + j];
      afr[ti] = fa.v;
    }
    // B 32x16 f16: same K order as A: lane half selects kbase (0/8); K=kbase..kbase+7, 16+kbase..16+kbase+7.
#pragma unroll
    for (int tj = 0; tj < 2; ++tj) {
      int nn  = wn + tj * 16 + l16;
      int kgu = half * 4;
      union { v16h v; uint32_t u[8]; } fb;
#pragma unroll
      for (int j = 0; j < 8; ++j) fb.u[j] = lB[nn][(j & 4) * 2 + kgu + (j & 3)];
      bfr[tj] = fb.v;
    }

#pragma unroll
    for (int ti = 0; ti < 4; ++ti)
#pragma unroll
      for (int tj = 0; tj < 2; ++tj)
        acc[ti][tj] = __builtin_amdgcn_wmma_f32_16x16x32_f16(
            false, afr[ti], false, bfr[tj], (short)0, acc[ti][tj], false, false);
    __syncthreads();
  }

  // Epilogue. C/D layout: vgpr v holds M = v + 8*half, N = lane&15.
#pragma unroll
  for (int tj = 0; tj < 2; ++tj) {
    int col = n0 + wn + tj * 16 + l16;
    float bv = bias ? bias[col] : 0.f;
#pragma unroll
    for (int ti = 0; ti < 4; ++ti) {
#pragma unroll
      for (int v = 0; v < 8; ++v) {
        int row = m0 + wm + ti * 16 + half * 8 + v;
        Cb[(long long)row * ldc + col] = acc[ti][tj][v] + bv;
      }
    }
  }
}

// ---------------------------------------------------------------------------
// Elementwise / support kernels (float4-vectorized; D_ % 4 == 0 throughout)
// ---------------------------------------------------------------------------

// positions = cumsum(tok != 0) * (tok != 0), serial per batch row (tiny).
__global__ void pos_kernel(const int* __restrict__ tok, int* __restrict__ pos,
                           int Bn, int Tlen) {
  int b = blockIdx.x * blockDim.x + threadIdx.x;
  if (b >= Bn) return;
  int acc = 0;
  for (int t = 0; t < Tlen; ++t) {
    int m = (tok[b * Tlen + t] != 0) ? 1 : 0;
    acc += m;
    pos[b * Tlen + t] = acc * m;
  }
}

__global__ void embed_kernel(const int* __restrict__ tok, const int* __restrict__ pos,
                             const float4* __restrict__ tokEmb,
                             const float4* __restrict__ posEmb,
                             float4* __restrict__ out, long long n4) {
  long long i = (long long)blockIdx.x * blockDim.x + threadIdx.x;
  if (i >= n4) return;
  const int D4 = D_ / 4;
  long long r = i / D4;
  int d4 = (int)(i % D4);
  float4 a = tokEmb[(long long)tok[r] * D4 + d4];
  float4 b = posEmb[(long long)pos[r] * D4 + d4];
  out[i] = make_float4(a.x + b.x, a.y + b.y, a.z + b.z, a.w + b.w);
}

// im2col for time-major activations X[(b*Tlen+t)*D + c]:
// out[r*CK + c*K + k] = X[b, t+k-padl, c] (0 outside range / masked src pad).
// One thread per (row, channel); writes the K_=3 taps. Tlen = 1<<tshift.
__global__ void im2col_kernel(const float* __restrict__ X,
                              const int* __restrict__ maskTok,
                              float* __restrict__ out,
                              int tshift, int padl, long long n /* rows*D */) {
  long long i = (long long)blockIdx.x * blockDim.x + threadIdx.x;
  if (i >= n) return;
  int c = (int)(i % D_);               // D_ = 512: pure shift/mask
  long long r = i / D_;
  const int Tlen = 1 << tshift;
  int b = (int)(r >> tshift);
  int t = (int)(r & (Tlen - 1));
  float* o = out + r * CK_ + (long long)c * K_;
  const float* xb = X + (((long long)b << tshift) * D_) + c;
#pragma unroll
  for (int k = 0; k < K_; ++k) {
    int tin = t + k - padl;
    float v = 0.f;
    if (tin >= 0 && tin < Tlen) {
      if (!(maskTok && maskTok[(b << tshift) + tin] == 0))
        v = xb[(long long)tin * D_];
    }
    o[k] = v;
  }
}

__global__ void glu_kernel(const float* __restrict__ in2c, float4* __restrict__ out,
                           long long n4) {
  long long i = (long long)blockIdx.x * blockDim.x + threadIdx.x;
  if (i >= n4) return;
  const int D4 = D_ / 4;
  long long r = i / D4;
  int c4 = (int)(i % D4);
  float4 a = *reinterpret_cast<const float4*>(in2c + r * C2_ + 4 * c4);
  float4 g = *reinterpret_cast<const float4*>(in2c + r * C2_ + D_ + 4 * c4);
  out[i] = make_float4(a.x / (1.f + __expf(-g.x)),
                       a.y / (1.f + __expf(-g.y)),
                       a.z / (1.f + __expf(-g.z)),
                       a.w / (1.f + __expf(-g.w)));
}

__global__ void addscale_kernel(const float4* __restrict__ a, const float4* __restrict__ b,
                                float4* __restrict__ out, long long n4, float s) {
  long long i = (long long)blockIdx.x * blockDim.x + threadIdx.x;
  if (i >= n4) return;
  float4 x = a[i], y = b[i];
  out[i] = make_float4((x.x + y.x) * s, (x.y + y.y) * s,
                       (x.z + y.z) * s, (x.w + y.w) * s);
}

// conved = where(pad, 0, conved); combined = (conved + emb) * R2
__global__ void mask_combine_kernel(float4* __restrict__ conved,
                                    const float4* __restrict__ emb,
                                    const int* __restrict__ srcTok,
                                    float4* __restrict__ combined, long long n4) {
  long long i = (long long)blockIdx.x * blockDim.x + threadIdx.x;
  if (i >= n4) return;
  const int D4 = D_ / 4;
  long long r = i / D4;
  float4 cv = conved[i];
  if (srcTok[r] == 0) cv = make_float4(0.f, 0.f, 0.f, 0.f);
  conved[i] = cv;
  float4 e = emb[i];
  combined[i] = make_float4((cv.x + e.x) * R2_, (cv.y + e.y) * R2_,
                            (cv.z + e.z) * R2_, (cv.w + e.w) * R2_);
}

// masked softmax over s (S_=256 == blockDim); one block per (b,t) row, in place.
__global__ void softmax_mask_kernel(float* __restrict__ energy,
                                    const int* __restrict__ srcTok) {
  __shared__ float red[256];
  int row = blockIdx.x;            // b*T + t
  int b = row / T_;
  int s = threadIdx.x;
  float e = energy[(long long)row * S_ + s];
  if (srcTok[b * S_ + s] == 0) e = -INFINITY;
  red[s] = e; __syncthreads();
  for (int off = 128; off > 0; off >>= 1) {
    if (s < off) red[s] = fmaxf(red[s], red[s + off]);
    __syncthreads();
  }
  float mx = red[0]; __syncthreads();
  float ex = (e == -INFINITY || !(mx > -INFINITY)) ? 0.f : __expf(e - mx);
  red[s] = ex; __syncthreads();
  for (int off = 128; off > 0; off >>= 1) {
    if (s < off) red[s] += red[s + off];
    __syncthreads();
  }
  float sum = red[0];
  energy[(long long)row * S_ + s] = (sum > 0.f) ? ex / sum : 0.f;
}

// ---------------------------------------------------------------------------
extern "C" void kernel_launch(void* const* d_in, const int* in_sizes, int n_in,
                              void* d_out, int out_size, void* d_ws, size_t ws_size,
                              hipStream_t stream) {
  (void)in_sizes; (void)n_in; (void)out_size; (void)ws_size;
  const int*   srcTok   = (const int*)d_in[0];
  const int*   trgTok   = (const int*)d_in[1];
  const float* encTokE  = (const float*)d_in[2];
  const float* encPosE  = (const float*)d_in[3];
  const float* decTokE  = (const float*)d_in[4];
  const float* decPosE  = (const float*)d_in[5];
  const float* encInW   = (const float*)d_in[6];
  const float* encInB   = (const float*)d_in[7];
  const float* encConvW = (const float*)d_in[8];
  const float* encConvB = (const float*)d_in[9];
  const float* encOutW  = (const float*)d_in[10];
  const float* encOutB  = (const float*)d_in[11];
  const float* decInW   = (const float*)d_in[12];
  const float* decInB   = (const float*)d_in[13];
  const float* decConvW = (const float*)d_in[14];
  const float* decConvB = (const float*)d_in[15];
  const float* attInW   = (const float*)d_in[16];
  const float* attInB   = (const float*)d_in[17];
  const float* attOutW  = (const float*)d_in[18];
  const float* attOutB  = (const float*)d_in[19];
  const float* decOutW  = (const float*)d_in[20];
  const float* decOutB  = (const float*)d_in[21];
  const float* vocabW   = (const float*)d_in[22];
  const float* vocabB   = (const float*)d_in[23];
  float* outLogits = (float*)d_out;

  const int BS = B_ * S_, BT = B_ * T_;

  float* ws = (float*)d_ws;
  int* posS = (int*)ws;              // 4096 ints
  int* posT = (int*)(ws + 4096);     // 2048 ints
  float* p = ws + 8192;
  float* encEmb = p; p += (size_t)BS * D_;
  float* encX   = p; p += (size_t)BS * D_;
  float* encCv  = p; p += (size_t)BS * D_;   // encoder conved
  float* encCb  = p; p += (size_t)BS * D_;   // encoder combined
  float* gtmp   = p; p += (size_t)BS * D_;   // glu scratch
  float* decEmb = p; p += (size_t)BT * D_;
  float* decX   = p; p += (size_t)BT * D_;
  float* decC   = p; p += (size_t)BT * D_;
  float* qbuf   = p; p += (size_t)BT * D_;
  float* aproj  = p; p += (size_t)BT * D_;
  float* attc   = p; p += (size_t)BT * D_;
  float* energy = p; p += (size_t)B_ * T_ * S_;
  float* imcol  = p; p += (size_t)BS * CK_;
  float* cvout  = p; p += (size_t)BS * C2_;

  auto gemm = [&](const float* A, int lda, long long sA,
                  const float* Bm, int ldb, long long sB,
                  float* Cm, int ldc, long long sC,
                  const float* bias, int M, int N, int Kd, int bT, int batch) {
    dim3 g((unsigned)(N / 128), (unsigned)(M / 128), (unsigned)batch);
    if (bT)
      wmma_gemm_kernel<1><<<g, 256, 0, stream>>>(A, lda, sA, Bm, ldb, sB,
                                                 Cm, ldc, sC, bias, M, N, Kd);
    else
      wmma_gemm_kernel<0><<<g, 256, 0, stream>>>(A, lda, sA, Bm, ldb, sB,
                                                 Cm, ldc, sC, bias, M, N, Kd);
  };
  auto blocks = [](long long n) { return dim3((unsigned)((n + 255) / 256)); };

  // ---- embeddings ----
  pos_kernel<<<1, 32, 0, stream>>>(srcTok, posS, B_, S_);
  pos_kernel<<<1, 32, 0, stream>>>(trgTok, posT, B_, T_);
  {
    long long n4 = (long long)BS * D_ / 4;
    embed_kernel<<<blocks(n4), 256, 0, stream>>>(
        srcTok, posS, (const float4*)encTokE, (const float4*)encPosE,
        (float4*)encEmb, n4);
  }
  {
    long long n4 = (long long)BT * D_ / 4;
    embed_kernel<<<blocks(n4), 256, 0, stream>>>(
        trgTok, posT, (const float4*)decTokE, (const float4*)decPosE,
        (float4*)decEmb, n4);
  }

  const long long nE4 = (long long)BS * D_ / 4;   // encoder [BS,D] float4 count
  const long long nD4 = (long long)BT * D_ / 4;   // decoder [BT,D] float4 count

  // ---- encoder ----
  gemm(encEmb, D_, 0, encInW, D_, 0, encX, D_, 0, encInB, BS, D_, D_, 0, 1);
  for (int l = 0; l < L_; ++l) {
    long long nIm = (long long)BS * D_;   // one thread per (row, channel)
    im2col_kernel<<<blocks(nIm), 256, 0, stream>>>(encX, srcTok, imcol,
                                                   8 /*S=256*/, K_ / 2, nIm);
    gemm(imcol, CK_, 0, encConvW + (size_t)l * C2_ * CK_, CK_, 0,
         cvout, C2_, 0, encConvB + (size_t)l * C2_, BS, C2_, CK_, 1, 1);
    glu_kernel<<<blocks(nE4), 256, 0, stream>>>(cvout, (float4*)gtmp, nE4);
    addscale_kernel<<<blocks(nE4), 256, 0, stream>>>(
        (const float4*)gtmp, (const float4*)encX, (float4*)encX, nE4, R2_);
  }
  gemm(encX, D_, 0, encOutW, D_, 0, encCv, D_, 0, encOutB, BS, D_, D_, 0, 1);
  mask_combine_kernel<<<blocks(nE4), 256, 0, stream>>>(
      (float4*)encCv, (const float4*)encEmb, srcTok, (float4*)encCb, nE4);

  // ---- decoder ----
  gemm(decEmb, D_, 0, decInW, D_, 0, decX, D_, 0, decInB, BT, D_, D_, 0, 1);
  for (int l = 0; l < L_; ++l) {
    long long nIm = (long long)BT * D_;
    im2col_kernel<<<blocks(nIm), 256, 0, stream>>>(decX, (const int*)nullptr, imcol,
                                                   7 /*T=128*/, K_ - 1, nIm); // causal
    gemm(imcol, CK_, 0, decConvW + (size_t)l * C2_ * CK_, CK_, 0,
         cvout, C2_, 0, decConvB + (size_t)l * C2_, BT, C2_, CK_, 1, 1);
    glu_kernel<<<blocks(nD4), 256, 0, stream>>>(cvout, (float4*)decC, nD4);

    // q = (c @ attn_in_w + b + emb) * R2
    gemm(decC, D_, 0, attInW + (size_t)l * D_ * D_, D_, 0, qbuf, D_, 0,
         attInB + (size_t)l * D_, BT, D_, D_, 0, 1);
    addscale_kernel<<<blocks(nD4), 256, 0, stream>>>(
        (const float4*)qbuf, (const float4*)decEmb, (float4*)qbuf, nD4, R2_);

    // energy[b,t,s] = q . enc_conved  (batched, B^T mode)
    gemm(qbuf, D_, (long long)T_ * D_, encCv, D_, (long long)S_ * D_,
         energy, S_, (long long)T_ * S_, nullptr, T_, S_, D_, 1, B_);
    softmax_mask_kernel<<<BT, 256, 0, stream>>>(energy, srcTok);

    // att = attn @ enc_combined  (batched)
    gemm(energy, S_, (long long)T_ * S_, encCb, D_, (long long)S_ * D_,
         attc, D_, (long long)T_ * D_, nullptr, T_, D_, S_, 0, B_);
    gemm(attc, D_, 0, attOutW + (size_t)l * D_ * D_, D_, 0, aproj, D_, 0,
         attOutB + (size_t)l * D_, BT, D_, D_, 0, 1);

    // c = (c + att) * R2 ; x = (c + res) * R2 ; res = x (in place)
    addscale_kernel<<<blocks(nD4), 256, 0, stream>>>(
        (const float4*)decC, (const float4*)aproj, (float4*)decC, nD4, R2_);
    addscale_kernel<<<blocks(nD4), 256, 0, stream>>>(
        (const float4*)decC, (const float4*)decX, (float4*)decX, nD4, R2_);
  }

  // ---- output projections ----
  gemm(decX, D_, 0, decOutW, D_, 0, decC, D_, 0, decOutB, BT, D_, D_, 0, 1);
  gemm(decC, D_, 0, vocabW, V_, 0, outLogits, V_, 0, vocabB, BT, V_, D_, 0, 1);
}
